// CapsuleNet_34316788695600
// MI455X (gfx1250) — matrix operand order repaired and checked by reading, not compiled, not run on hardware
//
#include <hip/hip_runtime.h>
#include <hip/hip_bf16.h>
#include <math.h>

typedef __attribute__((ext_vector_type(16))) _Float16 v16h;
typedef __attribute__((ext_vector_type(8)))  float    v8f;

#define BATCH 128
#define LDSROW 36   // padded row stride (floats): 144B, 16B-aligned, bank-spread

// LDS byte offset of a __shared__ location (ptrtoint of addrspace(3) pointer)
__device__ __forceinline__ unsigned ldsOffset(const void* p) {
    return (unsigned)(uintptr_t)(__attribute__((address_space(3))) const void*)p;
}
// async global->LDS copies (CDNA5 Tensor/Async path, tracked by ASYNCcnt)
__device__ __forceinline__ void asyncCopyB32(unsigned ldsOff, const float* g) {
    asm volatile("global_load_async_to_lds_b32 %0, %1, off"
                 :: "v"(ldsOff), "v"((unsigned long long)(uintptr_t)g) : "memory");
}
__device__ __forceinline__ void asyncCopyB128(unsigned ldsOff, const float* g) {
    asm volatile("global_load_async_to_lds_b128 %0, %1, off"
                 :: "v"(ldsOff), "v"((unsigned long long)(uintptr_t)g) : "memory");
}
__device__ __forceinline__ void waitAsync() {
    asm volatile("s_wait_asynccnt 0" ::: "memory");
}

// ---------------------------------------------------------------------------
// Implicit-GEMM convolution, f16 WMMA / f32 accumulate.
// NCHW fp32 activations, OIHW fp32 weights. 128 threads = 4 waves per block;
// block computes 128(M) x 16(N). The 16x32 weight tile is async-staged to LDS
// once per k-step and shared by all waves; each wave does two A fragments and
// two v_wmma per k-step. Requirements: M % 128 == 0, Cout % 16 == 0,
// OH/OW powers of two.
// ---------------------------------------------------------------------------
template<int KH, int KW, int PAD>
__global__ __launch_bounds__(128)
void conv2d_wmma(const float* __restrict__ in, const float* __restrict__ w,
                 const float* __restrict__ bias, float* __restrict__ out,
                 int Cin, int H, int W, int Cout,
                 int stride, int owshift, int ohshift, int relu, int wcount)
{
    __shared__ float Bs[16 * LDSROW];
    const int tid  = threadIdx.x;
    const int wave = tid >> 5;
    const int lane = tid & 31;
    const int half = lane >> 4;
    const int l16  = lane & 15;
    const int OW = 1 << owshift, OH = 1 << ohshift;
    const int HW = H * W;
    const int Ktot = Cin * KH * KW;

    // Per-lane A rows for this wave's two M-subtiles (shift decompose)
    int abase[2], ih0a[2], iw0a[2];
    #pragma unroll
    for (int sub = 0; sub < 2; ++sub) {
        const int m  = (blockIdx.x * 8 + wave * 2 + sub) * 16 + l16;
        const int ow = m & (OW - 1);
        const int oh = (m >> owshift) & (OH - 1);
        const int b  = m >> (owshift + ohshift);
        ih0a[sub] = oh * stride - PAD;
        iw0a[sub] = ow * stride - PAD;
        abase[sub] = b * Cin * HW + ih0a[sub] * W + iw0a[sub];
    }
    const int n = blockIdx.y * 16 + l16;   // B column consumed by this lane

    v8f acc0 = {}, acc1 = {};
    for (int k0 = 0; k0 < Ktot; k0 += 32) {
        // ---- async-stage the 16(n) x 32(k) fp32 weight tile into LDS ----
        #pragma unroll
        for (int j = 0; j < 4; ++j) {
            const int e   = tid + 128 * j;         // 512 elements
            const int row = e >> 5;                // n_local
            const int kk  = e & 31;
            int gidx = (blockIdx.y * 16 + row) * Ktot + k0 + kk;
            gidx = (gidx < wcount) ? gidx : (wcount - 1);   // stay in-bounds;
            // garbage past Ktot is harmless: A is exactly 0 at those k
            asyncCopyB32(ldsOffset(&Bs[row * LDSROW + kk]), w + gidx);
        }
        waitAsync();
        __syncthreads();

        // ---- B fragment from LDS: element t holds K = k0 + t + 16*half ----
        v16h bf;
        {
            const float4* __restrict__ Bs4 =
                (const float4*)(&Bs[l16 * LDSROW + 16 * half]);
            #pragma unroll
            for (int j = 0; j < 4; ++j) {
                const float4 q = Bs4[j];
                bf[4 * j + 0] = (_Float16)q.x;
                bf[4 * j + 1] = (_Float16)q.y;
                bf[4 * j + 2] = (_Float16)q.z;
                bf[4 * j + 3] = (_Float16)q.w;
            }
        }

        // ---- A fragments (im2col gather): K = k0 + t + 16*half ----
        v16h a0, a1;
        #pragma unroll
        for (int t = 0; t < 16; ++t) {
            const int ka = k0 + t + 16 * half;
            const bool okk = (ka < Ktot);
            int ci, kh_, kw_;
            if (KH == 1 && KW == 1) { ci = ka; kh_ = 0; kw_ = 0; }
            else {
                kw_ = ka % KW;                      // compile-time magic muls
                const int t2 = ka / KW;
                kh_ = t2 % KH;
                ci  = t2 / KH;
            }
            const int woff = ci * HW + kh_ * W + kw_;
            #pragma unroll
            for (int sub = 0; sub < 2; ++sub) {
                bool ok = okk;
                if (PAD > 0) {
                    const int ih = ih0a[sub] + kh_;
                    const int iw = iw0a[sub] + kw_;
                    ok = ok && (ih >= 0) && (ih < H) && (iw >= 0) && (iw < W);
                }
                const int off = abase[sub] + woff;
                float av = in[ok ? off : 0];        // branchless gather
                av = ok ? av : 0.0f;
                if (sub == 0) a0[t] = (_Float16)av; else a1[t] = (_Float16)av;
            }
        }

        acc0 = __builtin_amdgcn_wmma_f32_16x16x32_f16(false, a0, false, bf,
                                                      (short)0, acc0, false, false);
        acc1 = __builtin_amdgcn_wmma_f32_16x16x32_f16(false, a1, false, bf,
                                                      (short)0, acc1, false, false);
        __syncthreads();   // protect Bs before next-iteration overwrite
    }

    // ---- epilogue: D layout VGPR r -> M = r + 8*half, N = lane&15 ----
    const float bval = bias ? bias[n] : 0.0f;
    #pragma unroll
    for (int sub = 0; sub < 2; ++sub) {
        #pragma unroll
        for (int r = 0; r < 8; ++r) {
            const int mm = (blockIdx.x * 8 + wave * 2 + sub) * 16 + half * 8 + r;
            const int owo = mm & (OW - 1);
            const int oho = (mm >> owshift) & (OH - 1);
            const int bo  = mm >> (owshift + ohshift);
            float vout = (sub == 0 ? acc0[r] : acc1[r]) + bval;
            if (relu) vout = fmaxf(vout, 0.0f);
            out[(((size_t)bo * Cout + n) << (owshift + ohshift)) +
                (oho << owshift) + owo] = vout;
        }
    }
}

// ---------------------------------------------------------------------------
// GEMM: out[M,N] = act(A[M,K] * Wt[K,N] + bias); f16 WMMA / f32 accumulate.
// 128 threads = 4 waves; block computes 16(M) x 64(N). The 16x32 A-tile is
// async-staged (b128) into LDS and shared by all 4 waves; B loads are direct
// and lane-coalesced (Wt contiguous in n).
// Requirements: M % 16 == 0, N % 64 == 0, K % 32 == 0 (true for the decoder).
// act: 0 = none, 1 = relu, 2 = sigmoid
// ---------------------------------------------------------------------------
__global__ __launch_bounds__(128)
void gemm_wmma(const float* __restrict__ A, const float* __restrict__ Wt,
               const float* __restrict__ bias, float* __restrict__ out,
               int M, int K, int N, int act)
{
    __shared__ float As[16 * LDSROW];
    const int tid  = threadIdx.x;
    const int wave = tid >> 5;
    const int lane = tid & 31;
    const int half = lane >> 4;
    const int l16  = lane & 15;

    const int n = blockIdx.y * 64 + wave * 16 + l16;

    v8f acc = {};
    for (int k0 = 0; k0 < K; k0 += 32) {
        // ---- async-stage 16(m) x 32(k) A tile (one b128 per thread) ----
        {
            const int row = tid >> 3;              // 0..15
            const int kk  = (tid & 7) * 4;         // 0,4,...,28
            asyncCopyB128(ldsOffset(&As[row * LDSROW + kk]),
                          &A[(size_t)(blockIdx.x * 16 + row) * K + k0 + kk]);
        }
        waitAsync();
        __syncthreads();

        if (k0 + 32 < K)
            __builtin_prefetch(&Wt[(size_t)(k0 + 32) * N + n], 0, 1);

        // ---- A fragment from LDS: runs [8h,8h+8) and [16+8h,16+8h+8) ----
        v16h a;
        {
            const float4* __restrict__ As4 = (const float4*)(&As[l16 * LDSROW]);
            #pragma unroll
            for (int rseg = 0; rseg < 2; ++rseg) {
                const float4 q0 = As4[rseg * 4 + 2 * half + 0];
                const float4 q1 = As4[rseg * 4 + 2 * half + 1];
                a[8 * rseg + 0] = (_Float16)q0.x;
                a[8 * rseg + 1] = (_Float16)q0.y;
                a[8 * rseg + 2] = (_Float16)q0.z;
                a[8 * rseg + 3] = (_Float16)q0.w;
                a[8 * rseg + 4] = (_Float16)q1.x;
                a[8 * rseg + 5] = (_Float16)q1.y;
                a[8 * rseg + 6] = (_Float16)q1.z;
                a[8 * rseg + 7] = (_Float16)q1.w;
            }
        }

        // ---- B fragment: direct, lane-coalesced ----
        v16h bm;
        #pragma unroll
        for (int t = 0; t < 16; ++t) {
            const int kb = k0 + t + 8 * half + ((t >= 8) ? 8 : 0);
            bm[t] = (_Float16)Wt[(size_t)kb * N + n];
        }

        acc = __builtin_amdgcn_wmma_f32_16x16x32_f16(false, a, false, bm,
                                                     (short)0, acc, false, false);
        __syncthreads();
    }

    const float bval = bias ? bias[n] : 0.0f;
    #pragma unroll
    for (int r = 0; r < 8; ++r) {
        const int mm = blockIdx.x * 16 + half * 8 + r;
        float vout = acc[r] + bval;
        if (act == 1) vout = fmaxf(vout, 0.0f);
        else if (act == 2) vout = 1.0f / (1.0f + expf(-vout));
        out[(size_t)mm * N + n] = vout;
    }
}

// ---------------------------------------------------------------------------
// Squash primary capsules: p (B,128,4,4) NCHW -> u (B,256,8)
// ---------------------------------------------------------------------------
__global__ void squash_prims(const float* __restrict__ p, float* __restrict__ u, int total)
{
    const int idx = blockIdx.x * blockDim.x + threadIdx.x;
    if (idx >= total) return;
    const int nn = idx & 255;
    const int b  = idx >> 8;
    const int cap = nn >> 4;
    const int sp  = nn & 15;
    const int ph = sp >> 2, pw = sp & 3;
    float tv[8];
    float sn = 0.0f;
    #pragma unroll
    for (int d = 0; d < 8; ++d) {
        float x = p[(((size_t)b * 128 + cap * 8 + d) * 4 + ph) * 4 + pw];
        tv[d] = x;
        sn += x * x;
    }
    const float scale = sn / (1.0f + sn) / sqrtf(sn);
    #pragma unroll
    for (int d = 0; d < 8; ++d)
        u[((size_t)b * 256 + nn) * 8 + d] = tv[d] * scale;
}

// ---------------------------------------------------------------------------
// priors[b,c,n,o] = sum_i u[b,n,i] * route_w[c,n,i,o]  (K=8 -> VALU)
// ---------------------------------------------------------------------------
__global__ void compute_priors(const float* __restrict__ u, const float* __restrict__ rw,
                               float* __restrict__ priors, int total)
{
    const int idx = blockIdx.x * blockDim.x + threadIdx.x;
    if (idx >= total) return;
    const int n = idx & 255;
    const int cb = idx >> 8;          // b*10 + c
    const int c = cb % 10;
    const int b = cb / 10;
    float uv[8];
    #pragma unroll
    for (int i = 0; i < 8; ++i) uv[i] = u[((size_t)b * 256 + n) * 8 + i];
    const float* __restrict__ rwn = rw + ((size_t)c * 256 + n) * 128;
    #pragma unroll
    for (int o = 0; o < 16; ++o) {
        float s = 0.0f;
        #pragma unroll
        for (int i = 0; i < 8; ++i) s += uv[i] * rwn[i * 16 + o];
        priors[(((size_t)b * 10 + c) * 256 + n) * 16 + o] = s;
    }
}

// ---------------------------------------------------------------------------
// Dynamic routing, 3 iterations. One block (256 threads = routes) per (b,c).
// ---------------------------------------------------------------------------
__global__ __launch_bounds__(256)
void routing(const float* __restrict__ priors, float* __restrict__ v)
{
    const int bc = blockIdx.x;
    const int tid = threadIdx.x;
    __shared__ float red[256];
    __shared__ float outv[16];

    float pri[16];
    #pragma unroll
    for (int o = 0; o < 16; ++o)
        pri[o] = priors[((size_t)bc * 256 + tid) * 16 + o];

    float logit = 0.0f;
    for (int it = 0; it < 3; ++it) {
        red[tid] = logit; __syncthreads();
        for (int s = 128; s > 0; s >>= 1) {
            if (tid < s) red[tid] = fmaxf(red[tid], red[tid + s]);
            __syncthreads();
        }
        const float mx = red[0]; __syncthreads();
        const float e = expf(logit - mx);
        red[tid] = e; __syncthreads();
        for (int s = 128; s > 0; s >>= 1) {
            if (tid < s) red[tid] += red[tid + s];
            __syncthreads();
        }
        const float prob = e / red[0]; __syncthreads();

        for (int o = 0; o < 16; ++o) {
            red[tid] = prob * pri[o]; __syncthreads();
            for (int s = 128; s > 0; s >>= 1) {
                if (tid < s) red[tid] += red[tid + s];
                __syncthreads();
            }
            if (tid == 0) outv[o] = red[0];
            __syncthreads();
        }
        if (tid == 0) {
            float sn = 0.0f;
            for (int o = 0; o < 16; ++o) sn += outv[o] * outv[o];
            const float sc = sn / (1.0f + sn) / sqrtf(sn);
            for (int o = 0; o < 16; ++o) outv[o] *= sc;
        }
        __syncthreads();
        if (it < 2) {
            float d = 0.0f;
            #pragma unroll
            for (int o = 0; o < 16; ++o) d += pri[o] * outv[o];
            logit += d;
        }
    }
    if (tid < 16) v[(size_t)bc * 16 + tid] = outv[tid];
}

// ---------------------------------------------------------------------------
// classes = softmax(||v||), y = one-hot(argmax), d0 = (v*y).reshape(b,160)
// ---------------------------------------------------------------------------
__global__ void classify_mask(const float* __restrict__ v, float* __restrict__ classes,
                              float* __restrict__ d0)
{
    const int b = blockIdx.x * blockDim.x + threadIdx.x;
    if (b >= BATCH) return;
    float norms[10];
    float mx = -1e30f;
    for (int c = 0; c < 10; ++c) {
        float s = 0.0f;
        for (int o = 0; o < 16; ++o) {
            float x = v[((size_t)b * 10 + c) * 16 + o];
            s += x * x;
        }
        norms[c] = sqrtf(s);
        mx = fmaxf(mx, norms[c]);
    }
    float sum = 0.0f;
    for (int c = 0; c < 10; ++c) sum += expf(norms[c] - mx);
    int am = 0; float best = norms[0];
    for (int c = 1; c < 10; ++c) if (norms[c] > best) { best = norms[c]; am = c; }
    for (int c = 0; c < 10; ++c) {
        classes[b * 10 + c] = expf(norms[c] - mx) / sum;
        for (int o = 0; o < 16; ++o)
            d0[(size_t)b * 160 + c * 16 + o] =
                (c == am) ? v[((size_t)b * 10 + c) * 16 + o] : 0.0f;
    }
}

// ---------------------------------------------------------------------------
extern "C" void kernel_launch(void* const* d_in, const int* in_sizes, int n_in,
                              void* d_out, int out_size, void* d_ws, size_t ws_size,
                              hipStream_t stream) {
    (void)in_sizes; (void)n_in; (void)out_size; (void)ws_size;

    const float* x       = (const float*)d_in[0];
    const float* conv0_w = (const float*)d_in[1];
    const float* bn[4][6];
    for (int blk = 0; blk < 4; ++blk)
        for (int j = 0; j < 6; ++j)
            bn[blk][j] = (const float*)d_in[2 + blk * 6 + j];
    const float* prim_w  = (const float*)d_in[26];
    const float* prim_b  = (const float*)d_in[27];
    const float* route_w = (const float*)d_in[28];
    const float* dec_w[3] = { (const float*)d_in[29], (const float*)d_in[31], (const float*)d_in[33] };
    const float* dec_b[3] = { (const float*)d_in[30], (const float*)d_in[32], (const float*)d_in[34] };

    char* wsb = (char*)d_ws;
    float* actA = (float*)(wsb);                                 // 67,108,864 B
    float* actB = (float*)(wsb + (size_t)67108864);              // 134,217,728 B
    char*  extra = wsb + (size_t)67108864 + 134217728;
    float* u      = (float*)(extra);
    float* priors = (float*)(extra + (size_t)1048576);
    float* vcaps  = (float*)(extra + (size_t)1048576 + 20971520);
    float* d0     = (float*)(extra + (size_t)1048576 + 20971520 + 81920);
    float* h1     = (float*)(extra + (size_t)1048576 + 20971520 + 81920 + 81920);
    float* h2     = (float*)(extra + (size_t)1048576 + 20971520 + 81920 + 81920 + 262144);

    float* classes_out = (float*)d_out;
    float* dec_out     = (float*)d_out + 1280;

    auto shp = [](int v) { int s = 0; while ((1 << s) < v) ++s; return s; };

    #define CONV(KH, KW, PAD, in_, w_, b_, out_, Cin, H, W, Cout, S)                      \
        do {                                                                              \
            const int OH_ = (H + 2 * (PAD) - (KH)) / (S) + 1;                             \
            const int OW_ = (W + 2 * (PAD) - (KW)) / (S) + 1;                             \
            const int M_ = BATCH * OH_ * OW_;                                             \
            dim3 g_(M_ / 128, (Cout) / 16);                                               \
            conv2d_wmma<KH, KW, PAD><<<g_, dim3(128), 0, stream>>>(                       \
                in_, w_, b_, out_, Cin, H, W, Cout, S, shp(OW_), shp(OH_), relu_,         \
                (Cout) * (Cin) * (KH) * (KW));                                            \
        } while (0)

    int relu_ = 1;
    CONV(3, 3, 1, x,    conv0_w, nullptr, actA, 3,   128, 128, 32, 2);   // -> 64x64x32
    // block 1 (stride 1), 64x64
    CONV(1, 1, 0, actA, bn[0][0], bn[0][1], actB, 32, 64, 64, 16, 1);
    CONV(3, 3, 1, actB, bn[0][2], bn[0][3], actA, 16, 64, 64, 16, 1);
    CONV(1, 1, 0, actA, bn[0][4], bn[0][5], actB, 16, 64, 64, 64, 1);
    // block 2 (stride 2 at c3), 64x64 -> 32x32
    CONV(1, 1, 0, actB, bn[1][0], bn[1][1], actA, 64, 64, 64, 32, 1);
    CONV(3, 3, 1, actA, bn[1][2], bn[1][3], actB, 32, 64, 64, 32, 1);
    CONV(1, 1, 0, actB, bn[1][4], bn[1][5], actA, 32, 64, 64, 96, 2);
    // block 3 (stride 1), 32x32
    CONV(1, 1, 0, actA, bn[2][0], bn[2][1], actB, 96, 32, 32, 48, 1);
    CONV(3, 3, 1, actB, bn[2][2], bn[2][3], actA, 48, 32, 32, 48, 1);
    CONV(1, 1, 0, actA, bn[2][4], bn[2][5], actB, 48, 32, 32, 128, 1);
    // block 4 (stride 2 at c3), 32x32 -> 16x16
    CONV(1, 1, 0, actB, bn[3][0], bn[3][1], actA, 128, 32, 32, 64, 1);
    CONV(3, 3, 1, actA, bn[3][2], bn[3][3], actB, 64, 32, 32, 64, 1);
    CONV(1, 1, 0, actB, bn[3][4], bn[3][5], actA, 64, 32, 32, 160, 2);
    // primary caps conv: 160 -> 128, 9x9, stride 2, no pad, no relu -> (B,128,4,4)
    relu_ = 0;
    CONV(9, 9, 0, actA, prim_w, prim_b, actB, 160, 16, 16, 128, 2);
    #undef CONV

    squash_prims<<<dim3((BATCH * 256 + 255) / 256), dim3(256), 0, stream>>>(actB, u, BATCH * 256);
    compute_priors<<<dim3((BATCH * 10 * 256 + 255) / 256), dim3(256), 0, stream>>>(
        u, route_w, priors, BATCH * 10 * 256);
    routing<<<dim3(BATCH * 10), dim3(256), 0, stream>>>(priors, vcaps);
    classify_mask<<<dim3(1), dim3(BATCH), 0, stream>>>(vcaps, classes_out, d0);

    auto gemm = [&](const float* A, const float* Wt, const float* bias, float* out,
                    int M, int K, int N, int act) {
        dim3 g(M / 16, N / 64);
        gemm_wmma<<<g, dim3(128), 0, stream>>>(A, Wt, bias, out, M, K, N, act);
    };
    gemm(d0, dec_w[0], dec_b[0], h1, BATCH, 160, 512, 1);
    gemm(h1, dec_w[1], dec_b[1], h2, BATCH, 512, 1024, 1);
    gemm(h2, dec_w[2], dec_b[2], dec_out, BATCH, 1024, 49152, 2);
}
